// Attention_8117488190022
// MI455X (gfx1250) — matrix operand order repaired and checked by reading, not compiled, not run on hardware
//
#include <hip/hip_runtime.h>

// ---------------------------------------------------------------------------
// CDNA5 (gfx1250) wave32 WMMA types & helpers
// ---------------------------------------------------------------------------
typedef __attribute__((ext_vector_type(16))) __bf16 v16bf;
typedef __attribute__((ext_vector_type(8)))  __bf16 v8bf;
typedef __attribute__((ext_vector_type(8)))  float  v8f;

static __device__ __forceinline__ v16bf cat8(v8bf lo, v8bf hi) {
  return __builtin_shufflevector(lo, hi, 0,1,2,3,4,5,6,7,8,9,10,11,12,13,14,15);
}

static __device__ __forceinline__ v8f wmma_bf16(v16bf a, v16bf b, v8f c) {
  // D = A(16x32 bf16) * B(32x16 bf16) + C(16x16 f32)
  return __builtin_amdgcn_wmma_f32_16x16x32_bf16(false, a, false, b, (short)0, c,
                                                 false, false);
}

// ---------------------------------------------------------------------------
// One-shot f32 -> bf16 down-convert (8 elements / thread, b128 in, b128 out)
// ---------------------------------------------------------------------------
__global__ __launch_bounds__(256)
void cvt_f32_bf16_kernel(const float* __restrict__ s, __bf16* __restrict__ d,
                         int n) {
  const int i = (blockIdx.x * 256 + threadIdx.x) * 8;
  if (i >= n) return;
  const float4* sp = (const float4*)(s + i);
  const float4 a = sp[0], b = sp[1];
  v8bf o;
  o[0] = (__bf16)a.x; o[1] = (__bf16)a.y; o[2] = (__bf16)a.z; o[3] = (__bf16)a.w;
  o[4] = (__bf16)b.x; o[5] = (__bf16)b.y; o[6] = (__bf16)b.z; o[7] = (__bf16)b.w;
  *(v8bf*)(d + i) = o;
}

// ---------------------------------------------------------------------------
// Rank-8 LoRA projection, f32 input: XA[m][r] = sum_k X[m][k] * A[r][k]
// ---------------------------------------------------------------------------
__global__ __launch_bounds__(256)
void lora_xa_kernel(const float* __restrict__ X, const float* __restrict__ A,
                    float* __restrict__ XA, int M, int K) {
  const int idx = blockIdx.x * 256 + threadIdx.x;   // M*8 threads
  const int m = idx >> 3;
  const int r = idx & 7;
  const float4* xr = (const float4*)(X + (size_t)m * K);
  const float4* ar = (const float4*)(A + (size_t)r * K);
  float s = 0.0f;
  for (int k = 0; k < K / 4; ++k) {
    const float4 a = xr[k], b = ar[k];
    s += a.x * b.x + a.y * b.y + a.z * b.z + a.w * b.w;
  }
  XA[idx] = s;
}

// Same, bf16 activations (used for the attention output y).
__global__ __launch_bounds__(256)
void lora_xa_bf16_kernel(const __bf16* __restrict__ X, const float* __restrict__ A,
                         float* __restrict__ XA, int M, int K) {
  const int idx = blockIdx.x * 256 + threadIdx.x;
  const int m = idx >> 3;
  const int r = idx & 7;
  const v8bf*   xr = (const v8bf*)(X + (size_t)m * K);
  const float4* ar = (const float4*)(A + (size_t)r * K);
  float s = 0.0f;
  for (int k = 0; k < K / 8; ++k) {
    const v8bf  xv = xr[k];
    const float4 a0 = ar[2 * k], a1 = ar[2 * k + 1];
    s += (float)xv[0] * a0.x + (float)xv[1] * a0.y
       + (float)xv[2] * a0.z + (float)xv[3] * a0.w
       + (float)xv[4] * a1.x + (float)xv[5] * a1.y
       + (float)xv[6] * a1.z + (float)xv[7] * a1.w;
  }
  XA[idx] = s;
}

// ---------------------------------------------------------------------------
// LoRA GEMM (all-bf16 operands): Y[M,N] = X @ W^T + bias + XA @ LB^T
// One wave computes a 16x64 tile: 1 A-frag x 4 B-frags per K=32 step, and the
// fragments are straight 16-byte bf16 loads (no conversions in the hot loop).
// ---------------------------------------------------------------------------
__global__ __launch_bounds__(128)
void lora_gemm_kernel(const __bf16* __restrict__ X, const __bf16* __restrict__ W,
                      const float* __restrict__ bias, const float* __restrict__ XA,
                      const float* __restrict__ LB,
                      float* __restrict__ outF, __bf16* __restrict__ outB,
                      float* __restrict__ qout, float* __restrict__ kout,
                      int M, int N, int K, int Cdim) {
  const int lane = threadIdx.x & 31;
  const int wave = threadIdx.x >> 5;
  const int row  = lane & 15;
  const int hi   = lane >> 4;
  const int mbase = (blockIdx.y * 4 + wave) * 16;
  const int nbase = blockIdx.x * 64;

  // A-fragment: row = lane&15; K 0-15 (lanes 0-15) vs 16-31.
  const __bf16* xrow = X + (size_t)(mbase + row) * K + hi * 16;
  // B-fragment: col = lane&15; K 0-15 (lanes 0-15) vs 16-31. B[k][n] = W[n][k]
  // so each lane's 16 k-values are contiguous in the W row.
  const __bf16* wrow[4];
  #pragma unroll
  for (int t = 0; t < 4; ++t)
    wrow[t] = W + (size_t)(nbase + t * 16 + row) * K + hi * 16;

  v8f acc[4] = {};
  for (int k0 = 0; k0 < K; k0 += 32) {
    __builtin_prefetch(xrow + k0 + 128, 0, 0);      // global_prefetch_b8
    const v16bf a = cat8(*(const v8bf*)(xrow + k0),
                         *(const v8bf*)(xrow + k0 + 8));
    #pragma unroll
    for (int t = 0; t < 4; ++t) {
      const __bf16* wp = wrow[t] + k0;
      const v16bf bfrag = cat8(*(const v8bf*)wp, *(const v8bf*)(wp + 8));
      acc[t] = wmma_bf16(a, bfrag, acc[t]);
    }
  }

  // Epilogue: bias + rank-8 LoRA correction, then stores.
  float4 xa_lo[8], xa_hi[8];
  #pragma unroll
  for (int j = 0; j < 8; ++j) {
    const float4* xp4 = (const float4*)(XA + (size_t)(mbase + 8 * hi + j) * 8);
    xa_lo[j] = xp4[0];
    xa_hi[j] = xp4[1];
  }
  #pragma unroll
  for (int t = 0; t < 4; ++t) {
    const int n = nbase + t * 16 + row;
    const float4* lb4 = (const float4*)(LB + (size_t)n * 8);
    const float4 lb0 = lb4[0], lb1 = lb4[1];
    const float bn = bias[n];
    #pragma unroll
    for (int j = 0; j < 8; ++j) {
      const int gm = mbase + 8 * hi + j;
      float v = acc[t][j] + bn
              + xa_lo[j].x * lb0.x + xa_lo[j].y * lb0.y
              + xa_lo[j].z * lb0.z + xa_lo[j].w * lb0.w
              + xa_hi[j].x * lb1.x + xa_hi[j].y * lb1.y
              + xa_hi[j].z * lb1.z + xa_hi[j].w * lb1.w;
      if (outF) outF[(size_t)gm * N + n] = v;
      if (outB) outB[(size_t)gm * N + n] = (__bf16)v;
      if (qout) {
        if (n < Cdim)            qout[(size_t)gm * Cdim + n] = v;
        else if (n < 2 * Cdim)   kout[(size_t)gm * Cdim + (n - Cdim)] = v;
      }
    }
  }
}

// ---------------------------------------------------------------------------
// Causal flash attention: one wave per (b, h, 16-row q tile); 32-key blocks.
// qkv is bf16 [B, T, 3C]; y (bf16) is [B, T, C].  V tiles are staged into LDS
// with GLOBAL_LOAD_ASYNC_TO_LDS_B128 (ASYNCcnt) — no VGPR round trip.
// ---------------------------------------------------------------------------
__global__ __launch_bounds__(32)
void flash_attn_kernel(const __bf16* __restrict__ qkv, __bf16* __restrict__ y) {
  constexpr int T = 2048, C = 1024, D = 64;
  constexpr int S3C = 3 * C;
  __shared__ __bf16 lds_v[32 * 80];   // 32 keys x 64 d (stride 80 -> 160B rows)
  __shared__ __bf16 lds_p[16 * 32];   // P tile, accumulator-layout -> A-layout

  const int lane  = threadIdx.x & 31;
  const int row   = lane & 15;
  const int hi    = lane >> 4;
  const int qbase = blockIdx.x * 16;
  const int h     = blockIdx.y;
  const int b     = blockIdx.z;
  const size_t base = (size_t)b * T * S3C;

  // LDS byte offset of this lane's V row (generic-ptr low 32 bits = LDS offset)
  const unsigned vrow_lds = (unsigned)(uintptr_t)(lds_v + lane * 80);

  // Q fragments for the whole tile (d = 0..31 and 32..63)
  const __bf16* qrow = qkv + base + (size_t)(qbase + row) * S3C + h * D;
  const v16bf aq0 = cat8(*(const v8bf*)(qrow + hi * 16),
                         *(const v8bf*)(qrow + 8 + hi * 16));
  const v16bf aq1 = cat8(*(const v8bf*)(qrow + 32 + hi * 16),
                         *(const v8bf*)(qrow + 40 + hi * 16));

  v8f O[4] = {};
  float mstat[8], lstat[8];
  #pragma unroll
  for (int j = 0; j < 8; ++j) { mstat[j] = -__builtin_inff(); lstat[j] = 0.0f; }

  const float scale = 0.125f;          // 1/sqrt(64)
  const int kend = qbase + 16;         // keys 0 .. qbase+15
  for (int kb = 0; kb < kend; kb += 32) {
    // Async-stage V block (32 x 64 bf16) into LDS: one key row per lane,
    // 8 x b128; the instruction offset advances LDS and global together.
    {
      const unsigned long long vg =
          (unsigned long long)(uintptr_t)(qkv + base +
              (size_t)(kb + lane) * S3C + 2 * C + h * D);
      asm volatile(
        "global_load_async_to_lds_b128 %0, %1, off offset:0\n\t"
        "global_load_async_to_lds_b128 %0, %1, off offset:16\n\t"
        "global_load_async_to_lds_b128 %0, %1, off offset:32\n\t"
        "global_load_async_to_lds_b128 %0, %1, off offset:48\n\t"
        "global_load_async_to_lds_b128 %0, %1, off offset:64\n\t"
        "global_load_async_to_lds_b128 %0, %1, off offset:80\n\t"
        "global_load_async_to_lds_b128 %0, %1, off offset:96\n\t"
        "global_load_async_to_lds_b128 %0, %1, off offset:112"
        :: "v"(vrow_lds), "v"(vg) : "memory");
    }
    // Scores S(16x32) = Q(16x64) @ K^T : 4 WMMAs. K rows contiguous over d.
    v8f S0 = {}, S1 = {};
    {
      const __bf16* kp = qkv + base + (size_t)(kb + row) * S3C + C + h * D;
      v16bf bk;
      bk = cat8(*(const v8bf*)(kp + hi * 16), *(const v8bf*)(kp + hi * 16 + 8));
      S0 = wmma_bf16(aq0, bk, S0);
      bk = cat8(*(const v8bf*)(kp + 32 + hi * 16), *(const v8bf*)(kp + 40 + hi * 16));
      S0 = wmma_bf16(aq1, bk, S0);
      const __bf16* kp1 = kp + (size_t)16 * S3C;
      bk = cat8(*(const v8bf*)(kp1 + hi * 16), *(const v8bf*)(kp1 + hi * 16 + 8));
      S1 = wmma_bf16(aq0, bk, S1);
      bk = cat8(*(const v8bf*)(kp1 + 32 + hi * 16), *(const v8bf*)(kp1 + 40 + hi * 16));
      S1 = wmma_bf16(aq1, bk, S1);
    }
    // Online softmax. Row m = j + 8*hi lives in one 16-lane half -> shfl_xor<16.
    const bool need_mask = (kb + 31) > qbase;
    #pragma unroll
    for (int j = 0; j < 8; ++j) {
      float s0 = S0[j] * scale;
      float s1 = S1[j] * scale;
      if (need_mask) {
        const int qr = qbase + 8 * hi + j;
        if (kb + row      > qr) s0 = -1.0e30f;
        if (kb + 16 + row > qr) s1 = -1.0e30f;
      }
      float rm = fmaxf(s0, s1);
      #pragma unroll
      for (int off = 1; off < 16; off <<= 1) rm = fmaxf(rm, __shfl_xor(rm, off, 32));
      const float newm  = fmaxf(mstat[j], rm);
      const float alpha = __expf(mstat[j] - newm);
      mstat[j] = newm;
      const float e0 = __expf(s0 - newm);
      const float e1 = __expf(s1 - newm);
      float rs = e0 + e1;
      #pragma unroll
      for (int off = 1; off < 16; off <<= 1) rs += __shfl_xor(rs, off, 32);
      lstat[j] = lstat[j] * alpha + rs;
      #pragma unroll
      for (int t = 0; t < 4; ++t) O[t][j] *= alpha;
      lds_p[(8 * hi + j) * 32 + row]      = (__bf16)e0;
      lds_p[(8 * hi + j) * 32 + 16 + row] = (__bf16)e1;
    }
    // Wait for async V staging and the P-tile DS stores before LDS reads.
    asm volatile("s_wait_asynccnt 0\n\ts_wait_dscnt 0" ::: "memory");
    // P as A-fragment (16x32), V as B-fragments: O += P @ V, K=32 in one WMMA.
    const v16bf ap = cat8(*(const v8bf*)(lds_p + row * 32 + hi * 16),
                          *(const v8bf*)(lds_p + row * 32 + 8 + hi * 16));
    #pragma unroll
    for (int t = 0; t < 4; ++t) {
      v16bf bv;
      #pragma unroll
      for (int i = 0; i < 16; ++i) bv[i] = lds_v[(hi * 16 + i) * 80 + t * 16 + row];
      O[t] = wmma_bf16(ap, bv, O[t]);
    }
  }
  // Finalize: y[b, m, h*64 + d] = O / l  (bf16, feeds proj GEMM + xa2)
  #pragma unroll
  for (int j = 0; j < 8; ++j) {
    const float inv = 1.0f / lstat[j];
    const int gm = qbase + 8 * hi + j;
    __bf16* yr = y + ((size_t)b * T + gm) * C + h * D;
    #pragma unroll
    for (int t = 0; t < 4; ++t) yr[t * 16 + row] = (__bf16)(O[t][j] * inv);
  }
}

// ---------------------------------------------------------------------------
// Launch
// ---------------------------------------------------------------------------
extern "C" void kernel_launch(void* const* d_in, const int* in_sizes, int n_in,
                              void* d_out, int out_size, void* d_ws, size_t ws_size,
                              hipStream_t stream) {
  (void)in_sizes; (void)n_in; (void)out_size; (void)ws_size;
  const float* x   = (const float*)d_in[0];
  const float* w1  = (const float*)d_in[1];   // [3072, 1024]
  const float* b1  = (const float*)d_in[2];   // [3072]
  const float* A1  = (const float*)d_in[3];   // [8, 1024]
  const float* LB1 = (const float*)d_in[4];   // [3072, 8]
  const float* w2  = (const float*)d_in[5];   // [1024, 1024]
  const float* b2  = (const float*)d_in[6];   // [1024]
  const float* A2  = (const float*)d_in[7];   // [8, 1024]
  const float* LB2 = (const float*)d_in[8];   // [1024, 8]
  // d_in[9] = n_head (hardcoded 16)

  constexpr int Bb = 4, T = 2048, C = 1024;
  constexpr int M = Bb * T;                    // 8192 rows
  float* out = (float*)d_out;
  const size_t BTC = (size_t)M * C;

  // Workspace carve (bytes):
  char* ws = (char*)d_ws;
  size_t o = 0;
  __bf16* qkv = (__bf16*)(ws + o); o += (size_t)M * 3 * C * sizeof(__bf16); // 48MB
  __bf16* yb  = (__bf16*)(ws + o); o += (size_t)M * C * sizeof(__bf16);     // 16MB
  __bf16* xb  = (__bf16*)(ws + o); o += (size_t)M * C * sizeof(__bf16);     // 16MB
  __bf16* w1b = (__bf16*)(ws + o); o += (size_t)3 * C * C * sizeof(__bf16); //  6MB
  __bf16* w2b = (__bf16*)(ws + o); o += (size_t)C * C * sizeof(__bf16);     //  2MB
  float*  xa1 = (float*)(ws + o);  o += (size_t)M * 8 * sizeof(float);
  float*  xa2 = (float*)(ws + o);

  // 0) one-shot bf16 down-converts for WMMA operands
  cvt_f32_bf16_kernel<<<(M * C / 8) / 256, 256, 0, stream>>>(x, xb, M * C);
  cvt_f32_bf16_kernel<<<(3 * C * C / 8) / 256, 256, 0, stream>>>(w1, w1b, 3 * C * C);
  cvt_f32_bf16_kernel<<<(C * C / 8) / 256, 256, 0, stream>>>(w2, w2b, C * C);
  // 1) xa1 = x @ A1^T  (f32 precision on the rank-8 path)
  lora_xa_kernel<<<(M * 8) / 256, 256, 0, stream>>>(x, A1, xa1, M, C);
  // 2) qkv = x @ w1^T + b1 + xa1 @ LB1^T ; also emit query/key f32 outputs
  lora_gemm_kernel<<<dim3(3 * C / 64, M / 64), 128, 0, stream>>>(
      xb, w1b, b1, xa1, LB1, nullptr, qkv, out + BTC, out + 2 * BTC,
      M, 3 * C, C, C);
  // 3) y = causal softmax(q k^T / 8) v   (bf16 out)
  flash_attn_kernel<<<dim3(T / 16, 16, Bb), 32, 0, stream>>>(qkv, yb);
  // 4) xa2 = y @ A2^T
  lora_xa_bf16_kernel<<<(M * 8) / 256, 256, 0, stream>>>(yb, A2, xa2, M, C);
  // 5) out = y @ w2^T + b2 + xa2 @ LB2^T
  lora_gemm_kernel<<<dim3(C / 64, M / 64), 128, 0, stream>>>(
      yb, w2b, b2, xa2, LB2, out, nullptr, nullptr, nullptr,
      M, C, C, C);
}
